// BayerKP_70042326663875
// MI455X (gfx1250) — matrix operand order repaired and checked by reading, not compiled, not run on hardware
//
#include <hip/hip_runtime.h>

// ---------------------------------------------------------------------------
// CDNA5 (gfx1250) kernel-prediction demosaic.
// Conv layers 1..4: implicit-GEMM with v_wmma_f32_16x16x32_f16, f16 HWC
// activations, f32 accumulate. Input patches staged into LDS by the Tensor
// Data Mover (tensor_load_to_lds) with HW padding giving a conflict-free
// 72-half channel stride.
// ---------------------------------------------------------------------------

typedef __attribute__((ext_vector_type(16))) _Float16 v16h;
typedef __attribute__((ext_vector_type(8)))  _Float16 v8h;
typedef __attribute__((ext_vector_type(8)))  float    v8f;
typedef __attribute__((ext_vector_type(4)))  unsigned u32x4;
typedef __attribute__((ext_vector_type(4)))  int      i32x4;
typedef __attribute__((ext_vector_type(8)))  int      i32x8;

#if defined(__has_builtin)
#if __has_builtin(__builtin_amdgcn_tensor_load_to_lds) && \
    __has_builtin(__builtin_amdgcn_s_wait_tensorcnt)
#define HAVE_TDM 1
#endif
#endif

#define QH 384              // quarter-res plane size
#define QP (QH*QH)          // 147456 elements per plane

// ------------------------- prep: gray sum + phase split ---------------------
__global__ __launch_bounds__(256) void prep_kernel(
    const float* __restrict__ mosaic,   // (2,3,768,768)
    float* __restrict__ planes)         // [batch][4][384*384] : g0, b, r, g1
{
    size_t gid = (size_t)blockIdx.x * 256 + threadIdx.x;
    size_t total = (size_t)2 * QP;
    if (gid >= total) return;
    int batch = (int)(gid / QP);
    int p = (int)(gid % QP);
    int y = p / QH, x = p % QH;
    const float* mb = mosaic + (size_t)batch * 3 * 768 * 768;
    auto gray = [&](int yy, int xx) {
        size_t o = (size_t)yy * 768 + xx;
        return mb[o] + mb[589824 + o] + mb[1179648 + o];
    };
    float* pb = planes + (size_t)batch * 4 * QP;
    int o = y * QH + x;
    pb[o]            = gray(2*y,   2*x  );   // g0
    pb[QP + o]       = gray(2*y+1, 2*x  );   // b_
    pb[2*QP + o]     = gray(2*y,   2*x+1);   // r
    pb[3*QP + o]     = gray(2*y+1, 2*x+1);   // g1
}

// ------------------- weight repack into WMMA A-fragment order ---------------
// dst layout: [mtile][kt(18)][lane(32)][16 halfs]; K = (kh*3+kw)*64 + ci.
// 16-bit A 16x32 layout (contiguous, matches B): lane<16 -> K{0..15}, lane>=16 -> K{16..31}.
__global__ __launch_bounds__(256) void repack_kernel(
    const float* __restrict__ W,        // (Cout, 64, 3, 3) OIHW f32
    _Float16* __restrict__ dst,
    int mtiles, int CoutReal)
{
    size_t gid = (size_t)blockIdx.x * 256 + threadIdx.x;
    size_t total = (size_t)mtiles * 18 * 32 * 16;
    if (gid >= total) return;
    int j    = (int)(gid & 15);
    int lane = (int)((gid >> 4) & 31);
    int kt   = (int)((gid >> 9) % 18);
    int mt   = (int)(gid / (18 * 512));
    int m    = mt * 16 + (lane & 15);
    int klo  = (lane >> 4) * 16;
    int kl   = j + klo;
    int k    = kt * 32 + kl;
    int t    = k >> 6, ci = k & 63;
    int kh   = t / 3, kw = t - kh * 3;
    float v = 0.f;
    if (m < CoutReal)
        v = W[((size_t)m * 64 + ci) * 9 + kh * 3 + kw];
    dst[gid] = (_Float16)v;
}

// ----------------------- conv0: 4 -> 64, direct VALU ------------------------
__global__ __launch_bounds__(256) void conv0_kernel(
    const float* __restrict__ planes,   // [batch][4][384*384]
    const float* __restrict__ W0,       // (64,4,3,3)
    const float* __restrict__ b0,
    _Float16* __restrict__ out,         // [batch][382*382][64] HWC f16
    size_t outBatchStride)              // in elements
{
    __shared__ float w[64 * 36];
    for (int i = threadIdx.x; i < 64 * 36; i += 256) w[i] = W0[i];
    __syncthreads();

    size_t gid = (size_t)blockIdx.x * 256 + threadIdx.x;
    const int HO = 382, WO = 382;
    size_t total = (size_t)2 * HO * WO;
    if (gid >= total) return;
    int batch = (int)(gid / ((size_t)HO * WO));
    int p = (int)(gid % ((size_t)HO * WO));
    int y = p / WO, x = p % WO;
    const float* pb = planes + (size_t)batch * 4 * QP;

    float xin[36];
#pragma unroll
    for (int ci = 0; ci < 4; ci++)
#pragma unroll
        for (int kh = 0; kh < 3; kh++)
#pragma unroll
            for (int kw = 0; kw < 3; kw++)
                xin[ci * 9 + kh * 3 + kw] = pb[ci * QP + (y + kh) * QH + (x + kw)];

    _Float16* op = out + (size_t)batch * outBatchStride + ((size_t)y * WO + x) * 64;
#pragma unroll 2
    for (int cg = 0; cg < 8; cg++) {
        union { uint4 u; _Float16 h[8]; } pk;
#pragma unroll
        for (int j = 0; j < 8; j++) {
            int co = cg * 8 + j;
            float a = b0[co];
            const float* wr = w + co * 36;
#pragma unroll
            for (int k = 0; k < 36; k++) a = fmaf(wr[k], xin[k], a);
            pk.h[j] = (_Float16)fmaxf(a, 0.f);
        }
        ((uint4*)op)[cg] = pk.u;
    }
}

// ------------------- conv 3x3, Cin=64, implicit GEMM WMMA -------------------
// Grid: x = width tiles of 128 px, y = output row, z = batch*nchunks.
// 128 threads = 4 waves; each wave: 2 N-tiles (32 px) x MTILES M-tiles.
// Input patch (3 x 130 x 64ch f16) staged into LDS by the TDM with HW padding
// (pad 4 dwords every 32 dwords -> channel stride 72 halfs, conflict-free).
template <int MTILES, bool RELU>
__global__ __launch_bounds__(128, 1) void conv3x3_wmma(
    const _Float16* __restrict__ in, int Win,
    _Float16* __restrict__ out, int Wout, int CoutStride, int CoutReal,
    const _Float16* __restrict__ Apack,   // packed fragments
    const float* __restrict__ bias,
    int mbase, int nchunks,
    size_t inBatchStride, size_t outBatchStride)
{
    const int CP = 72;                        // LDS channel stride (halfs)
    __shared__ _Float16 smem[3 * 130 * CP];   // 56,160 B

    int bz = blockIdx.z;
    int batch = bz / nchunks;
    int chunk = bz - batch * nchunks;
    int mb = mbase + chunk * 64;              // output-channel base for this WG

    in  += (size_t)batch * inBatchStride;
    out += (size_t)batch * outBatchStride;

    int x0 = blockIdx.x * 128;
    int y  = blockIdx.y;
    int tid = threadIdx.x;

#if HAVE_TDM
    // ---- Tensor Data Mover: 2D tile (3 rows) x (130 px * 64 ch halfs) ----
    if (tid < 32) {                           // one wave issues the DMA
        unsigned long long ga =
            (unsigned long long)(size_t)(in + ((size_t)y * Win + x0) * 64);
        unsigned lds_base = (unsigned)(size_t)(void*)smem;  // low 32 = LDS offset
        unsigned dim0 = (unsigned)((Win - x0) * 64);        // OOB cols -> zero
        unsigned str0 = (unsigned)(Win * 64);               // row stride (units)

        u32x4 g0;
        g0.x = 1u;                                          // count=1
        g0.y = lds_base;                                    // lds_addr
        g0.z = (unsigned)(ga & 0xFFFFFFFFu);                // global_addr[31:0]
        g0.w = (unsigned)((ga >> 32) & 0x01FFFFFFu) | (2u << 30);  // [56:32]|type=2

        i32x8 g1;
        g1[0] = (int)((1u << 16) |              // data_size = 2 bytes
                      (1u << 20) |              // pad_enable
                      (4u << 22) |              // pad_interval: every 32 dwords
                      (3u << 25));              // pad_amount: 4 dwords
        g1[1] = (int)((dim0 & 0xFFFFu) << 16);  // tensor_dim0[15:0]
        g1[2] = (int)((dim0 >> 16) | (3u << 16));      // dim0[31:16] | tensor_dim1=3
        g1[3] = (int)(8320u << 16);             // tile_dim0 = 130*64
        g1[4] = 3;                              // tile_dim1 = 3, tile_dim2 = 0
        g1[5] = (int)str0;                      // tensor_dim0_stride[31:0]
        g1[6] = 0;                              // stride0[47:32] | stride1[15:0]
        g1[7] = 0;

        i32x4 z4 = {0, 0, 0, 0};
#if __clang_major__ >= 23
        i32x8 z8 = {0, 0, 0, 0, 0, 0, 0, 0};
        __builtin_amdgcn_tensor_load_to_lds(g0, g1, z4, z4, z8, 0);
#else
        __builtin_amdgcn_tensor_load_to_lds(g0, g1, z4, z4, 0);
#endif
        __builtin_amdgcn_s_wait_tensorcnt(0);
    }
#else
    // ---- fallback: manual staged copy (clamped cols) ----
    for (int i = tid; i < 3 * 130 * 8; i += 128) {
        int ck = i & 7;
        int c  = (i >> 3) % 130;
        int rr = i / (130 * 8);
        int sx = x0 + c; if (sx > Win - 1) sx = Win - 1;
        const uint4* src = (const uint4*)(in + ((size_t)(y + rr) * Win + sx) * 64) + ck;
        *((uint4*)(smem + (rr * 130 + c) * CP + ck * 8)) = *src;
    }
#endif
    __syncthreads();

    int wave = tid >> 5, lane = tid & 31;
    int ln = lane & 15, hi = lane >> 4;
    int colbase = wave * 32 + ln;             // first of two 16-px N-tiles

    v8f acc[2][MTILES];
#pragma unroll
    for (int n = 0; n < 2; n++)
#pragma unroll
        for (int m = 0; m < MTILES; m++)
#pragma unroll
            for (int j = 0; j < 8; j++) acc[n][m][j] = 0.f;

#pragma unroll
    for (int kt = 0; kt < 18; ++kt) {
        int t  = kt >> 1;
        int kh = t / 3, kw = t - kh * 3;
        int chb = (kt & 1) * 32 + hi * 16;    // dense-B: 16 consecutive K per lane
        const _Float16* bp0 = smem + ((kh * 130) + colbase + kw) * CP + chb;
        const _Float16* bp1 = bp0 + 16 * CP;
        v16h bf0, bf1;
        *((v8h*)&bf0)       = *((const v8h*)bp0);
        *(((v8h*)&bf0) + 1) = *((const v8h*)(bp0 + 8));
        *((v8h*)&bf1)       = *((const v8h*)bp1);
        *(((v8h*)&bf1) + 1) = *((const v8h*)(bp1 + 8));
#pragma unroll
        for (int m = 0; m < MTILES; m++) {
            const v16h* ap = (const v16h*)(Apack +
                (((size_t)((mb >> 4) + m) * 18 + kt) * 32 + lane) * 16);
            v16h af = *ap;                    // shared by both N-tiles
            acc[0][m] = __builtin_amdgcn_wmma_f32_16x16x32_f16(
                false, af, false, bf0, (short)0, acc[0][m], false, false);
            acc[1][m] = __builtin_amdgcn_wmma_f32_16x16x32_f16(
                false, af, false, bf1, (short)0, acc[1][m], false, false);
        }
    }

    // ---- epilogue: bias (+ReLU), f32 -> f16, packed b128 stores ----
#pragma unroll
    for (int n = 0; n < 2; n++) {
        int px = x0 + wave * 32 + n * 16 + ln;
        if (px < Wout) {
            size_t pixoff = ((size_t)y * Wout + px) * CoutStride;
#pragma unroll
            for (int m = 0; m < MTILES; m++) {
                int c0 = mb + m * 16 + hi * 8;
                union { uint4 u; _Float16 h[8]; } pk;
#pragma unroll
                for (int j = 0; j < 8; j++) {
                    float bb = (c0 + j < CoutReal) ? bias[c0 + j] : 0.f;
                    float v = acc[n][m][j] + bb;
                    if (RELU) v = fmaxf(v, 0.f);
                    pk.h[j] = (_Float16)v;
                }
                *((uint4*)(out + pixoff + c0)) = pk.u;   // pad chans exist in buffer
            }
        }
    }
}

// --------------- fused softmax + 7x7 kernel-apply + 2x2 unroll --------------
__global__ __launch_bounds__(256) void sa_kernel(
    const _Float16* __restrict__ kern,  // [batch][374*374][496]
    const float* __restrict__ planes,   // [batch][4][384*384]
    float* __restrict__ outp)           // (2,3,748,748)
{
    size_t gid = (size_t)blockIdx.x * 256 + threadIdx.x;
    const int HO = 374, WO = 374;
    size_t per = (size_t)HO * WO;
    if (gid >= 2 * per) return;
    int batch = (int)(gid / per);
    int p = (int)(gid % per);
    int y = p / WO, x = p % WO;

    const _Float16* kp = kern + ((size_t)batch * per + p) * 496;
    const float* pb  = planes + (size_t)batch * 4 * QP;
    const float* g0p = pb;
    const float* bp  = pb + QP;
    const float* rp  = pb + 2 * QP;
    const float* g1p = pb + 3 * QP;
    int patchbase = (y + 2) * QH + (x + 2);

    auto apply49 = [&](const float* plane, int cb) -> float {
        float mx = -1e30f;
        for (int i = 0; i < 49; i++) mx = fmaxf(mx, (float)kp[cb + i]);
        float s = 0.f, acc = 0.f;
        for (int i = 0; i < 49; i++) {
            float e = __expf((float)kp[cb + i] - mx);
            s += e;
            acc = fmaf(e, plane[patchbase + (i / 7) * QH + (i % 7)], acc);
        }
        return acc / s;
    };
    auto apply98 = [&](int cb) -> float {   // joint softmax over g0 / g1 halves
        float mx = -1e30f;
        for (int i = 0; i < 98; i++) mx = fmaxf(mx, (float)kp[cb + i]);
        float s = 0.f, acc = 0.f;
        for (int i = 0; i < 98; i++) {
            float e = __expf((float)kp[cb + i] - mx);
            s += e;
            const float* pl = (i < 49) ? g0p : g1p;
            int ii = (i < 49) ? i : i - 49;
            acc = fmaf(e, pl[patchbase + (ii / 7) * QH + (ii % 7)], acc);
        }
        return acc / s;
    };

    int center = (y + 5) * QH + (x + 5);
    float r_c0 = apply49(rp, 0);
    float r_c2 = apply49(rp, 49);
    float r_c3 = apply49(rp, 98);
    float r_c1 = rp[center];
    float b_c0 = apply49(bp, 147);
    float b_c1 = apply49(bp, 196);
    float b_c3 = apply49(bp, 245);
    float b_c2 = bp[center];
    float g_c1 = apply98(294);
    float g_c2 = apply98(392);
    float g_c0 = g0p[center];
    float g_c3 = g1p[center];

    size_t ob = (size_t)batch * 3 * 748 * 748;
    float* oR = outp + ob;
    float* oG = oR + (size_t)748 * 748;
    float* oB = oG + (size_t)748 * 748;
    size_t r0 = (size_t)(2 * y) * 748 + 2 * x;
    oR[r0] = r_c0; oR[r0 + 1] = r_c1; oR[r0 + 748] = r_c2; oR[r0 + 749] = r_c3;
    oG[r0] = g_c0; oG[r0 + 1] = g_c1; oG[r0 + 748] = g_c2; oG[r0 + 749] = g_c3;
    oB[r0] = b_c0; oB[r0 + 1] = b_c1; oB[r0 + 748] = b_c2; oB[r0 + 749] = b_c3;
}

// ---------------------------------------------------------------------------
extern "C" void kernel_launch(void* const* d_in, const int* in_sizes, int n_in,
                              void* d_out, int out_size, void* d_ws, size_t ws_size,
                              hipStream_t stream)
{
    (void)in_sizes; (void)n_in; (void)out_size; (void)ws_size;
    const float* mosaic = (const float*)d_in[0];
    const float* W0 = (const float*)d_in[1];  const float* b0 = (const float*)d_in[2];
    const float* W1 = (const float*)d_in[3];  const float* b1 = (const float*)d_in[4];
    const float* W2 = (const float*)d_in[5];  const float* b2 = (const float*)d_in[6];
    const float* W3 = (const float*)d_in[7];  const float* b3 = (const float*)d_in[8];
    const float* W4 = (const float*)d_in[9];  const float* b4 = (const float*)d_in[10];
    float* outp = (float*)d_out;

    // Workspace carve-out (~358 MB total).
    char* ws = (char*)d_ws;
    size_t off = 0;
    auto alloc = [&](size_t b) { size_t o = off; off = (off + b + 255) & ~(size_t)255; return o; };
    float*    planes = (float*)   (ws + alloc((size_t)2 * 4 * QP * 4));
    const size_t ACT_BS = (size_t)382 * 382 * 64;             // elements per batch
    _Float16* actA   = (_Float16*)(ws + alloc(2 * ACT_BS * 2));
    _Float16* actB   = (_Float16*)(ws + alloc(2 * ACT_BS * 2));
    const size_t KERN_BS = (size_t)374 * 374 * 496;
    _Float16* kern   = (_Float16*)(ws + alloc(2 * KERN_BS * 2));
    _Float16* pw1    = (_Float16*)(ws + alloc((size_t)4  * 18 * 512 * 2));
    _Float16* pw2    = (_Float16*)(ws + alloc((size_t)4  * 18 * 512 * 2));
    _Float16* pw3    = (_Float16*)(ws + alloc((size_t)4  * 18 * 512 * 2));
    _Float16* pw4    = (_Float16*)(ws + alloc((size_t)31 * 18 * 512 * 2));

    dim3 blk256(256), blk128(128);
    prep_kernel<<<(2 * QP + 255) / 256, blk256, 0, stream>>>(mosaic, planes);
    repack_kernel<<<(4 * 18 * 512 + 255) / 256, blk256, 0, stream>>>(W1, pw1, 4, 64);
    repack_kernel<<<(4 * 18 * 512 + 255) / 256, blk256, 0, stream>>>(W2, pw2, 4, 64);
    repack_kernel<<<(4 * 18 * 512 + 255) / 256, blk256, 0, stream>>>(W3, pw3, 4, 64);
    repack_kernel<<<(31 * 18 * 512 + 255) / 256, blk256, 0, stream>>>(W4, pw4, 31, 490);

    conv0_kernel<<<(2 * 382 * 382 + 255) / 256, blk256, 0, stream>>>(planes, W0, b0, actA, ACT_BS);

    // conv1..3: 64->64, ReLU
    conv3x3_wmma<4, true ><<<dim3(3, 380, 2), blk128, 0, stream>>>(
        actA, 382, actB, 380, 64, 64, pw1, b1, 0, 1, ACT_BS, ACT_BS);
    conv3x3_wmma<4, true ><<<dim3(3, 378, 2), blk128, 0, stream>>>(
        actB, 380, actA, 378, 64, 64, pw2, b2, 0, 1, ACT_BS, ACT_BS);
    conv3x3_wmma<4, true ><<<dim3(3, 376, 2), blk128, 0, stream>>>(
        actA, 378, actB, 376, 64, 64, pw3, b3, 0, 1, ACT_BS, ACT_BS);

    // conv4: 64->490 (padded 496), no ReLU. 7 chunks x 64ch + 1 chunk x 48ch.
    conv3x3_wmma<4, false><<<dim3(3, 374, 14), blk128, 0, stream>>>(
        actB, 376, kern, 374, 496, 490, pw4, b4, 0, 7, ACT_BS, KERN_BS);
    conv3x3_wmma<3, false><<<dim3(3, 374, 2), blk128, 0, stream>>>(
        actB, 376, kern, 374, 496, 490, pw4, b4, 448, 1, ACT_BS, KERN_BS);

    sa_kernel<<<(2 * 374 * 374 + 255) / 256, blk256, 0, stream>>>(kern, planes, outp);
}
